// MMDecoderDoubleBranch_19774029431325
// MI455X (gfx1250) — matrix operand order repaired and checked by reading, not compiled, not run on hardware
//
#include <hip/hip_runtime.h>

typedef __attribute__((ext_vector_type(16))) __bf16 v16bf;
typedef __attribute__((ext_vector_type(8)))  float  v8f;

// ---------------- geometry ----------------
#define kB 32768
#define kH 128
#define kM 6
#define kT 30
#define ROWS 32           // batch rows per block (2 WMMA M-tiles)
#define NTHREADS 256      // 8 waves

#define HSP 136   // bf16 stride for ROWSx128 LDS tiles (bank-conflict padding)
#define XSP 40    // bf16 stride for ROWSx32 xin tile
#define GSP 520   // bf16 stride for ROWSx512 gates
#define TSP 132   // f32 stride for ROWSx128 f32 tiles

// bf16-element offsets of prepacked weight fragments in d_ws
#define OFF_WIH 0
#define OFF_WHH (OFF_WIH + 32 * 512)              // 16384
#define OFF_MW1 (OFF_WHH + 32 * 4 * 512)          // 81920
#define OFF_MW2 (OFF_MW1 + 6 * 8 * 4 * 512)       // 180224
#define OFF_CW1 (OFF_MW2 + 6 * 8 * 4 * 512)       // 278528
#define OFF_CW2 (OFF_CW1 + 8 * 4 * 512)           // 294912
#define FR_TOTAL (OFF_CW2 + 8 * 4 * 512)          // 311296 bf16 elems

// ---------------- helpers ----------------
// native bf16 converts (single v_cvt ops on gfx1250, RNE)
__device__ __forceinline__ unsigned short f2bf(float f) {
  __bf16 b = (__bf16)f;
  return __builtin_bit_cast(unsigned short, b);
}
__device__ __forceinline__ float bf2f(unsigned short u) {
  return (float)__builtin_bit_cast(__bf16, u);
}

__device__ __forceinline__ float rcp_fast(float x) { return __builtin_amdgcn_rcpf(x); }
__device__ __forceinline__ float sigmoidf_(float x) { return rcp_fast(1.f + __expf(-x)); }
// tanh(x) = 1 - 2/(exp(2x)+1); saturates correctly at +-1
__device__ __forceinline__ float tanhf_(float x) {
  return 1.f - 2.f * rcp_fast(__expf(2.f * x) + 1.f);
}
__device__ __forceinline__ float seluf_(float x) {
  const float l = 1.0507009873554805f, a = 1.6732632423543772f;
  return x > 0.f ? l * x : l * a * (__expf(x) - 1.f);
}

// A-fragment (16x32 bf16, MxK) gathered from an LDS tile stored row-major.
// Layout per ISA 7.12.2: VGPR v, lane half: K = (v>>2)*16 + half*8 + (v&3)*2 (+pair)
// (compiler merges the 8 dword reads into two ds_load_b128)
__device__ __forceinline__ v16bf load_a_frag(const unsigned short* sh, int stride,
                                             int kc, int r0) {
  int lane = threadIdx.x & 31;
  int hf = lane >> 4, row = r0 + (lane & 15);
  union { v16bf v; unsigned u[8]; } a;
#pragma unroll
  for (int vv = 0; vv < 8; ++vv) {
    int kl = ((vv >> 2) << 4) + (hf << 3) + ((vv & 3) << 1);
    a.u[vv] = *(const unsigned*)(sh + row * stride + (kc << 5) + kl);
  }
  return a.v;
}

// B-fragment (32x16 bf16, KxN) prepacked lane-major: lane's 16 values contiguous.
__device__ __forceinline__ v16bf load_b_frag(const unsigned short* g) {
  int lane = threadIdx.x & 31;
  union { v16bf v; uint4 q[2]; } t;
  const uint4* p = (const uint4*)(g + lane * 16);
  t.q[0] = p[0];
  t.q[1] = p[1];
  return t.v;
}

// C/D (16x16 f32): VGPR v -> row M = v + 8*(lane>>4), col N = lane&15
__device__ __forceinline__ void store_acc(float* dst, int stride, int r0, int n0, v8f acc) {
  int lane = threadIdx.x & 31;
  int hf = lane >> 4, n = lane & 15;
#pragma unroll
  for (int vv = 0; vv < 8; ++vv)
    dst[(r0 + vv + (hf << 3)) * stride + n0 + n] = acc[vv];
}
__device__ __forceinline__ void store_acc_bf(unsigned short* dst, int stride, int r0,
                                             int n0, v8f acc) {
  int lane = threadIdx.x & 31;
  int hf = lane >> 4, n = lane & 15;
#pragma unroll
  for (int vv = 0; vv < 8; ++vv)
    dst[(r0 + vv + (hf << 3)) * stride + n0 + n] = f2bf(acc[vv]);
}

#define WMMA_BF16(A, B, C) \
  __builtin_amdgcn_wmma_f32_16x16x32_bf16(false, (A), false, (B), (short)0, (C), false, false)

// ---------------- weight repack kernel ----------------
// Converts f32 weights into bf16 WMMA B-fragments (lane-major) in d_ws.
// Within a 512-elem fragment: e = lane*16 + i ; K_local = (i>>3)*16 + (lane>>4)*8 + (i&7) ; N_local = lane&15
__global__ __launch_bounds__(256) void prep_kernel(
    const float* __restrict__ W_ih, const float* __restrict__ W_hh,
    const float* __restrict__ mW1, const float* __restrict__ mW2,
    const float* __restrict__ cW1, const float* __restrict__ cW2,
    unsigned short* __restrict__ frags) {
  int idx = blockIdx.x * 256 + threadIdx.x;
  if (idx >= FR_TOTAL) return;
  float val = 0.f;
  if (idx < OFF_WHH) {                        // W_ih^T (16x512), K padded to 32
    int i2 = idx - OFF_WIH;
    int tile = i2 >> 9, e = i2 & 511;
    int lane = e >> 4, i = e & 15;
    int k = ((i >> 3) << 4) + ((lane >> 4) << 3) + (i & 7);
    int n = (tile << 4) + (lane & 15);
    val = (k < 16) ? W_ih[n * 16 + k] : 0.f;
  } else if (idx < OFF_MW1) {                 // W_hh^T (128x512)
    int i2 = idx - OFF_WHH;
    int tile = i2 >> 11, chunk = (i2 >> 9) & 3, e = i2 & 511;
    int lane = e >> 4, i = e & 15;
    int k = (chunk << 5) + ((i >> 3) << 4) + ((lane >> 4) << 3) + (i & 7);
    int n = (tile << 4) + (lane & 15);
    val = W_hh[n * kH + k];
  } else if (idx < OFF_MW2) {                 // mW1 (6,128,128)
    int i2 = idx - OFF_MW1;
    int m = i2 >> 14, r = i2 & 16383;
    int tile = r >> 11, chunk = (r >> 9) & 3, e = r & 511;
    int lane = e >> 4, i = e & 15;
    int k = (chunk << 5) + ((i >> 3) << 4) + ((lane >> 4) << 3) + (i & 7);
    int n = (tile << 4) + (lane & 15);
    val = mW1[m * kH * kH + k * kH + n];
  } else if (idx < OFF_CW1) {                 // mW2 (6,128,128)
    int i2 = idx - OFF_MW2;
    int m = i2 >> 14, r = i2 & 16383;
    int tile = r >> 11, chunk = (r >> 9) & 3, e = r & 511;
    int lane = e >> 4, i = e & 15;
    int k = (chunk << 5) + ((i >> 3) << 4) + ((lane >> 4) << 3) + (i & 7);
    int n = (tile << 4) + (lane & 15);
    val = mW2[m * kH * kH + k * kH + n];
  } else if (idx < OFF_CW2) {                 // cW1 (128,128)
    int i2 = idx - OFF_CW1;
    int tile = i2 >> 11, chunk = (i2 >> 9) & 3, e = i2 & 511;
    int lane = e >> 4, i = e & 15;
    int k = (chunk << 5) + ((i >> 3) << 4) + ((lane >> 4) << 3) + (i & 7);
    int n = (tile << 4) + (lane & 15);
    val = cW1[k * kH + n];
  } else {                                    // cW2 (128,128)
    int i2 = idx - OFF_CW2;
    int tile = i2 >> 11, chunk = (i2 >> 9) & 3, e = i2 & 511;
    int lane = e >> 4, i = e & 15;
    int k = (chunk << 5) + ((i >> 3) << 4) + ((lane >> 4) << 3) + (i & 7);
    int n = (tile << 4) + (lane & 15);
    val = cW2[k * kH + n];
  }
  frags[idx] = f2bf(val);
}

// ---------------- fused persistent decoder kernel ----------------
// One block = 32 batch rows (2 WMMA M-tiles) held in LDS across all T steps; 8 waves.
// Each wave loads a B fragment once and issues it against both M-tiles (2x L2 reuse).
__global__ __launch_bounds__(NTHREADS, 1) void decoder_kernel(
    const float* __restrict__ last_obs_rel,
    const float* __restrict__ h0, const float* __restrict__ c0,
    const float* __restrict__ W_se, const float* __restrict__ b_se,
    const float* __restrict__ b_ih, const float* __restrict__ b_hh,
    const float* __restrict__ mg1w, const float* __restrict__ mg1b,
    const float* __restrict__ mg2w, const float* __restrict__ mg2b,
    const float* __restrict__ mWo, const float* __restrict__ mbo,
    const float* __restrict__ cg1w, const float* __restrict__ cg1b,
    const float* __restrict__ cg2w, const float* __restrict__ cg2b,
    const float* __restrict__ cWo, const float* __restrict__ cbo,
    const unsigned short* __restrict__ frags,
    float* __restrict__ pred, float* __restrict__ conf) {
  __shared__ unsigned short sh_h[ROWS * HSP];   // h (bf16: WMMA A + residuals)
  __shared__ unsigned short sh_x[ROWS * XSP];   // xin (bf16, K padded to 32)
  __shared__ float sh_c[ROWS * kH];             // c (f32)
  __shared__ float sh_bias[4 * kH];             // b_ih + b_hh
  __shared__ float sh_u[8320];                  // union: gates bf16 | (t f32, t bf16, rel)

  unsigned short* sh_gb = (unsigned short*)sh_u;              // ROWS x GSP bf16 (33280 B)
  float* sh_t = sh_u;                                         // ROWS x TSP f32 (16896 B)
  unsigned short* sh_tb = (unsigned short*)(sh_u + ROWS * TSP);  // ROWS x HSP bf16
  float* sh_rel = sh_u + ROWS * TSP + (ROWS * HSP) / 2;       // ROWS x 12 f32

  const int tid = threadIdx.x;
  const int wave = __builtin_amdgcn_readfirstlane(tid >> 5);  // uniform -> no EXEC masking
  const int b0 = blockIdx.x * ROWS;

  // ---- phase 0: load state, biases, initial dec_in ----
  for (int idx = tid; idx < ROWS * kH; idx += NTHREADS) {
    int r = idx >> 7, j = idx & 127;
    float h = h0[(size_t)(b0 + r) * kH + j];
    sh_c[idx] = c0[(size_t)(b0 + r) * kH + j];
    sh_h[r * HSP + j] = f2bf(h);
  }
  for (int j = tid; j < 4 * kH; j += NTHREADS) sh_bias[j] = b_ih[j] + b_hh[j];
  for (int idx = tid; idx < ROWS * 16; idx += NTHREADS) {
    int r = idx >> 4, e = idx & 15;
    float rx = last_obs_rel[(b0 + r) * 2 + 0];
    float ry = last_obs_rel[(b0 + r) * 2 + 1];
    float acc = b_se[e];
#pragma unroll
    for (int m = 0; m < kM; ++m)
      acc += rx * W_se[e * 12 + 2 * m] + ry * W_se[e * 12 + 2 * m + 1];
    sh_x[r * XSP + e] = f2bf(seluf_(acc));
    sh_x[r * XSP + 16 + e] = 0;   // zero K-padding
  }
  __syncthreads();

  for (int t = 0; t < kT; ++t) {
    // ---- phase 1: gates = [xin|h] @ [W_ih|W_hh]^T  (32x512) ----
    {
      v16bf ax[2], ah[2][4];
#pragma unroll
      for (int mt = 0; mt < 2; ++mt) {
        ax[mt] = load_a_frag(sh_x, XSP, 0, mt * 16);
#pragma unroll
        for (int kc = 0; kc < 4; ++kc) ah[mt][kc] = load_a_frag(sh_h, HSP, kc, mt * 16);
      }
      for (int n = wave; n < 32; n += 8) {
        v16bf bx = load_b_frag(frags + OFF_WIH + n * 512);
        v16bf bh0 = load_b_frag(frags + OFF_WHH + (n * 4 + 0) * 512);
        v16bf bh1 = load_b_frag(frags + OFF_WHH + (n * 4 + 1) * 512);
        v16bf bh2 = load_b_frag(frags + OFF_WHH + (n * 4 + 2) * 512);
        v16bf bh3 = load_b_frag(frags + OFF_WHH + (n * 4 + 3) * 512);
#pragma unroll
        for (int mt = 0; mt < 2; ++mt) {
          v8f acc = {};
          acc = WMMA_BF16(ax[mt], bx, acc);
          acc = WMMA_BF16(ah[mt][0], bh0, acc);
          acc = WMMA_BF16(ah[mt][1], bh1, acc);
          acc = WMMA_BF16(ah[mt][2], bh2, acc);
          acc = WMMA_BF16(ah[mt][3], bh3, acc);
          store_acc_bf(sh_gb, GSP, mt * 16, n << 4, acc);
        }
      }
    }
    __syncthreads();

    // ---- phase 2: LSTM pointwise (gate order i,f,g,o) ----
    for (int idx = tid; idx < ROWS * kH; idx += NTHREADS) {
      int r = idx >> 7, j = idx & 127;
      float gi = bf2f(sh_gb[r * GSP + j]) + sh_bias[j];
      float gf = bf2f(sh_gb[r * GSP + kH + j]) + sh_bias[kH + j];
      float gc = bf2f(sh_gb[r * GSP + 2 * kH + j]) + sh_bias[2 * kH + j];
      float go = bf2f(sh_gb[r * GSP + 3 * kH + j]) + sh_bias[3 * kH + j];
      float c = sigmoidf_(gf) * sh_c[idx] + sigmoidf_(gi) * tanhf_(gc);
      float h = sigmoidf_(go) * tanhf_(c);
      sh_c[idx] = c;
      sh_h[r * HSP + j] = f2bf(h);
    }
    __syncthreads();

    // h A-fragments shared by all mode heads this step
    v16bf ahm[2][4];
#pragma unroll
    for (int mt = 0; mt < 2; ++mt)
#pragma unroll
      for (int kc = 0; kc < 4; ++kc) ahm[mt][kc] = load_a_frag(sh_h, HSP, kc, mt * 16);

    for (int m = 0; m < kM; ++m) {
      // GEMM1: t = h @ mW1[m]; wave owns n-tile = wave, both M-tiles
      {
        const unsigned short* wb = frags + OFF_MW1 + (size_t)((m * 8 + wave) * 4) * 512;
        v16bf bb0 = load_b_frag(wb + 0 * 512);
        v16bf bb1 = load_b_frag(wb + 1 * 512);
        v16bf bb2 = load_b_frag(wb + 2 * 512);
        v16bf bb3 = load_b_frag(wb + 3 * 512);
#pragma unroll
        for (int mt = 0; mt < 2; ++mt) {
          v8f acc = {};
          acc = WMMA_BF16(ahm[mt][0], bb0, acc);
          acc = WMMA_BF16(ahm[mt][1], bb1, acc);
          acc = WMMA_BF16(ahm[mt][2], bb2, acc);
          acc = WMMA_BF16(ahm[mt][3], bb3, acc);
          store_acc(sh_t, TSP, mt * 16, wave << 4, acc);
        }
      }
      __syncthreads();

      // GN1 (+ scale/bias) + ReLU -> bf16 (8 lanes per row, shuffle reduce)
      {
        int r = tid >> 3, s = tid & 7;
        float sm = 0.f, sq = 0.f;
#pragma unroll
        for (int jj = 0; jj < 16; ++jj) {
          float v = sh_t[r * TSP + s * 16 + jj];
          sm += v; sq += v * v;
        }
#pragma unroll
        for (int dl = 1; dl < 8; dl <<= 1) {
          sm += __shfl_xor(sm, dl, 8);
          sq += __shfl_xor(sq, dl, 8);
        }
        float mu = sm * (1.f / 128.f);
        float rstd = rsqrtf(sq * (1.f / 128.f) - mu * mu + 1e-5f);
#pragma unroll
        for (int jj = 0; jj < 16; ++jj) {
          int j = s * 16 + jj;
          float v = (sh_t[r * TSP + j] - mu) * rstd * mg1w[m * kH + j] + mg1b[m * kH + j];
          sh_tb[r * HSP + j] = f2bf(fmaxf(v, 0.f));
        }
      }
      __syncthreads();

      // GEMM2: t = relu1 @ mW2[m]
      {
        v16bf at[2][4];
#pragma unroll
        for (int mt = 0; mt < 2; ++mt)
#pragma unroll
          for (int kc = 0; kc < 4; ++kc) at[mt][kc] = load_a_frag(sh_tb, HSP, kc, mt * 16);
        const unsigned short* wb = frags + OFF_MW2 + (size_t)((m * 8 + wave) * 4) * 512;
        v16bf bb0 = load_b_frag(wb + 0 * 512);
        v16bf bb1 = load_b_frag(wb + 1 * 512);
        v16bf bb2 = load_b_frag(wb + 2 * 512);
        v16bf bb3 = load_b_frag(wb + 3 * 512);
#pragma unroll
        for (int mt = 0; mt < 2; ++mt) {
          v8f acc = {};
          acc = WMMA_BF16(at[mt][0], bb0, acc);
          acc = WMMA_BF16(at[mt][1], bb1, acc);
          acc = WMMA_BF16(at[mt][2], bb2, acc);
          acc = WMMA_BF16(at[mt][3], bb3, acc);
          store_acc(sh_t, TSP, mt * 16, wave << 4, acc);
        }
      }
      __syncthreads();

      // GN2 + residual(h) + ReLU fused with output head (t @ mWo + mbo):
      // after the stats, each 8-lane group holds its full row in registers,
      // so the 128-long dot is done as per-lane partials + width-8 reduce.
      {
        int r = tid >> 3, s = tid & 7;
        float sm = 0.f, sq = 0.f;
#pragma unroll
        for (int jj = 0; jj < 16; ++jj) {
          float v = sh_t[r * TSP + s * 16 + jj];
          sm += v; sq += v * v;
        }
#pragma unroll
        for (int dl = 1; dl < 8; dl <<= 1) {
          sm += __shfl_xor(sm, dl, 8);
          sq += __shfl_xor(sq, dl, 8);
        }
        float mu = sm * (1.f / 128.f);
        float rstd = rsqrtf(sq * (1.f / 128.f) - mu * mu + 1e-5f);
        float p0 = 0.f, p1 = 0.f;
#pragma unroll
        for (int jj = 0; jj < 16; ++jj) {
          int j = s * 16 + jj;
          float v = (sh_t[r * TSP + j] - mu) * rstd * mg2w[m * kH + j] + mg2b[m * kH + j];
          v = fmaxf(v + bf2f(sh_h[r * HSP + j]), 0.f);
          p0 += v * mWo[(m * kH + j) * 2 + 0];
          p1 += v * mWo[(m * kH + j) * 2 + 1];
        }
#pragma unroll
        for (int dl = 1; dl < 8; dl <<= 1) {
          p0 += __shfl_xor(p0, dl, 8);
          p1 += __shfl_xor(p1, dl, 8);
        }
        if (s == 0) {
          float o0 = p0 + mbo[m * 2 + 0];
          float o1 = p1 + mbo[m * 2 + 1];
          size_t pb = (((size_t)(b0 + r) * kM + m) * kT + t) * 2;
          pred[pb + 0] = o0;
          pred[pb + 1] = o1;
          sh_rel[r * 12 + m * 2 + 0] = o0;
          sh_rel[r * 12 + m * 2 + 1] = o1;
        }
      }
      __syncthreads();
    }

    // ---- next xin = selu(rel @ W_se^T + b_se) ----
    for (int idx = tid; idx < ROWS * 16; idx += NTHREADS) {
      int r = idx >> 4, e = idx & 15;
      float acc = b_se[e];
#pragma unroll
      for (int j = 0; j < 12; ++j) acc += sh_rel[r * 12 + j] * W_se[e * 12 + j];
      sh_x[r * XSP + e] = f2bf(seluf_(acc));
    }
    __syncthreads();
  }

  // ---- confidence head on hT ----
  {
    v16bf ahm[2][4];
#pragma unroll
    for (int mt = 0; mt < 2; ++mt)
#pragma unroll
      for (int kc = 0; kc < 4; ++kc) ahm[mt][kc] = load_a_frag(sh_h, HSP, kc, mt * 16);
    {
      const unsigned short* wb = frags + OFF_CW1 + (wave * 4) * 512;
      v16bf bb0 = load_b_frag(wb + 0 * 512);
      v16bf bb1 = load_b_frag(wb + 1 * 512);
      v16bf bb2 = load_b_frag(wb + 2 * 512);
      v16bf bb3 = load_b_frag(wb + 3 * 512);
#pragma unroll
      for (int mt = 0; mt < 2; ++mt) {
        v8f acc = {};
        acc = WMMA_BF16(ahm[mt][0], bb0, acc);
        acc = WMMA_BF16(ahm[mt][1], bb1, acc);
        acc = WMMA_BF16(ahm[mt][2], bb2, acc);
        acc = WMMA_BF16(ahm[mt][3], bb3, acc);
        store_acc(sh_t, TSP, mt * 16, wave << 4, acc);
      }
    }
    __syncthreads();
    {  // GN(cg1) + relu -> bf16
      int r = tid >> 3, s = tid & 7;
      float sm = 0.f, sq = 0.f;
#pragma unroll
      for (int jj = 0; jj < 16; ++jj) {
        float v = sh_t[r * TSP + s * 16 + jj];
        sm += v; sq += v * v;
      }
#pragma unroll
      for (int dl = 1; dl < 8; dl <<= 1) {
        sm += __shfl_xor(sm, dl, 8);
        sq += __shfl_xor(sq, dl, 8);
      }
      float mu = sm * (1.f / 128.f);
      float rstd = rsqrtf(sq * (1.f / 128.f) - mu * mu + 1e-5f);
#pragma unroll
      for (int jj = 0; jj < 16; ++jj) {
        int j = s * 16 + jj;
        float v = (sh_t[r * TSP + j] - mu) * rstd * cg1w[j] + cg1b[j];
        sh_tb[r * HSP + j] = f2bf(fmaxf(v, 0.f));
      }
    }
    __syncthreads();
    {
      v16bf at[2][4];
#pragma unroll
      for (int mt = 0; mt < 2; ++mt)
#pragma unroll
        for (int kc = 0; kc < 4; ++kc) at[mt][kc] = load_a_frag(sh_tb, HSP, kc, mt * 16);
      const unsigned short* wb = frags + OFF_CW2 + (wave * 4) * 512;
      v16bf bb0 = load_b_frag(wb + 0 * 512);
      v16bf bb1 = load_b_frag(wb + 1 * 512);
      v16bf bb2 = load_b_frag(wb + 2 * 512);
      v16bf bb3 = load_b_frag(wb + 3 * 512);
#pragma unroll
      for (int mt = 0; mt < 2; ++mt) {
        v8f acc = {};
        acc = WMMA_BF16(at[mt][0], bb0, acc);
        acc = WMMA_BF16(at[mt][1], bb1, acc);
        acc = WMMA_BF16(at[mt][2], bb2, acc);
        acc = WMMA_BF16(at[mt][3], bb3, acc);
        store_acc(sh_t, TSP, mt * 16, wave << 4, acc);
      }
    }
    __syncthreads();
    {  // GN(cg2) + residual(hT) + relu fused with logits (t @ cWo + cbo)
      int r = tid >> 3, s = tid & 7;
      float sm = 0.f, sq = 0.f;
#pragma unroll
      for (int jj = 0; jj < 16; ++jj) {
        float v = sh_t[r * TSP + s * 16 + jj];
        sm += v; sq += v * v;
      }
#pragma unroll
      for (int dl = 1; dl < 8; dl <<= 1) {
        sm += __shfl_xor(sm, dl, 8);
        sq += __shfl_xor(sq, dl, 8);
      }
      float mu = sm * (1.f / 128.f);
      float rstd = rsqrtf(sq * (1.f / 128.f) - mu * mu + 1e-5f);
      float pl[6] = {0.f, 0.f, 0.f, 0.f, 0.f, 0.f};
#pragma unroll
      for (int jj = 0; jj < 16; ++jj) {
        int j = s * 16 + jj;
        float v = (sh_t[r * TSP + j] - mu) * rstd * cg2w[j] + cg2b[j];
        v = fmaxf(v + bf2f(sh_h[r * HSP + j]), 0.f);
#pragma unroll
        for (int mm = 0; mm < kM; ++mm) pl[mm] += v * cWo[j * kM + mm];
      }
#pragma unroll
      for (int mm = 0; mm < kM; ++mm)
#pragma unroll
        for (int dl = 1; dl < 8; dl <<= 1) pl[mm] += __shfl_xor(pl[mm], dl, 8);
      if (s == 0) {
#pragma unroll
        for (int mm = 0; mm < kM; ++mm) sh_rel[r * 6 + mm] = pl[mm] + cbo[mm];
      }
    }
    __syncthreads();
    if (tid < ROWS) {  // softmax over modes
      float mx = sh_rel[tid * 6];
#pragma unroll
      for (int mm = 1; mm < 6; ++mm) mx = fmaxf(mx, sh_rel[tid * 6 + mm]);
      float e[6], s = 0.f;
#pragma unroll
      for (int mm = 0; mm < 6; ++mm) { e[mm] = __expf(sh_rel[tid * 6 + mm] - mx); s += e[mm]; }
      float inv = rcp_fast(s);
#pragma unroll
      for (int mm = 0; mm < 6; ++mm) conf[(size_t)(b0 + tid) * 6 + mm] = e[mm] * inv;
    }
  }
}

// ---------------- launch ----------------
extern "C" void kernel_launch(void* const* d_in, const int* in_sizes, int n_in,
                              void* d_out, int out_size, void* d_ws, size_t ws_size,
                              hipStream_t stream) {
  (void)in_sizes; (void)n_in; (void)out_size; (void)ws_size;
  const float* last_obs_rel = (const float*)d_in[1];
  const float* h0   = (const float*)d_in[2];
  const float* c0   = (const float*)d_in[3];
  const float* W_se = (const float*)d_in[4];
  const float* b_se = (const float*)d_in[5];
  const float* W_ih = (const float*)d_in[6];
  const float* W_hh = (const float*)d_in[7];
  const float* b_ih = (const float*)d_in[8];
  const float* b_hh = (const float*)d_in[9];
  const float* mW1  = (const float*)d_in[10];
  const float* mg1w = (const float*)d_in[11];
  const float* mg1b = (const float*)d_in[12];
  const float* mW2  = (const float*)d_in[13];
  const float* mg2w = (const float*)d_in[14];
  const float* mg2b = (const float*)d_in[15];
  const float* mWo  = (const float*)d_in[16];
  const float* mbo  = (const float*)d_in[17];
  const float* cW1  = (const float*)d_in[18];
  const float* cg1w = (const float*)d_in[19];
  const float* cg1b = (const float*)d_in[20];
  const float* cW2  = (const float*)d_in[21];
  const float* cg2w = (const float*)d_in[22];
  const float* cg2b = (const float*)d_in[23];
  const float* cWo  = (const float*)d_in[24];
  const float* cbo  = (const float*)d_in[25];

  unsigned short* frags = (unsigned short*)d_ws;
  float* pred = (float*)d_out;
  float* conf = pred + (size_t)kB * kM * kT * 2;

  prep_kernel<<<(FR_TOTAL + 255) / 256, 256, 0, stream>>>(
      W_ih, W_hh, mW1, mW2, cW1, cW2, frags);

  decoder_kernel<<<kB / ROWS, NTHREADS, 0, stream>>>(
      last_obs_rel, h0, c0, W_se, b_se, b_ih, b_hh,
      mg1w, mg1b, mg2w, mg2b, mWo, mbo,
      cg1w, cg1b, cg2w, cg2b, cWo, cbo,
      frags, pred, conf);
}
